// AttentionMixer_19739669692571
// MI455X (gfx1250) — compile-verified, matrix operand order checked by reading, not run on hardware
//
#include <hip/hip_runtime.h>
#include <hip/hip_bf16.h>
#include <math.h>

// ---------------- problem constants ----------------
#define Bc   4
#define NQ   2048
#define NK   4096
#define Ec   256
#define DPC  128
#define Hc   4
#define HD   64          // head dim
#define LOG2E 1.4426950408889634f

typedef __attribute__((ext_vector_type(16))) _Float16 v16h;
typedef __attribute__((ext_vector_type(8)))  float    v8f;
typedef __attribute__((ext_vector_type(4)))  unsigned v4u;
typedef __attribute__((ext_vector_type(8)))  int      v8i;
typedef __attribute__((ext_vector_type(4)))  int      v4i;

__device__ __forceinline__ v8f wmma_f16(v16h a, v16h b, v8f c) {
    return __builtin_amdgcn_wmma_f32_16x16x32_f16(
        /*neg_a=*/false, a, /*neg_b=*/false, b,
        /*c_mod=*/(short)0, c, /*reuse_a=*/false, /*reuse_b=*/false);
}

__device__ __forceinline__ v16h cvt16(const float* p) {
    v16h r;
#pragma unroll
    for (int j = 0; j < 16; ++j) r[j] = (_Float16)p[j];
    return r;
}

// ------------------------------------------------------------
// TDM: async-load a 32x64 f16 tile (rows = keys, 64 contiguous dims)
// from global (row stride Ec halfs) into LDS at lds_off, via a 2D D#.
// Wave-uniform arguments -> SGPR descriptor groups. Tracked by TENSORcnt.
// ------------------------------------------------------------
__device__ __forceinline__ void tdm_load_k_tile(const _Float16* gp, unsigned lds_off) {
    unsigned long long ga = (unsigned long long)(uintptr_t)gp;
    v4u g0;
    g0[0] = 1u;                                  // count=1, user descriptor
    g0[1] = lds_off;                             // LDS byte address
    g0[2] = (unsigned)ga;                        // global addr [31:0]
    g0[3] = (unsigned)((ga >> 32) & 0x1FFFFFFull)// global addr [56:32]
            | (2u << 30);                        // type = 2 ("image")
    v8i g1 = {
        0x00010000,                              // wg_mask=0, data_size=1 (2B)
        0,                                       // atomic barrier addr = 0
        1,                                       // tensor_dim0 = 0x10000 (hi bits)
        (64 << 16) | 1,                          // tile_dim0=64 | tensor_dim1 hi
        32,                                      // tile_dim1=32, tile_dim2=0
        Ec,                                      // tensor_dim0_stride = 256
        0, 0
    };
    v4i gz = {0, 0, 0, 0};                       // groups 2/3 unused (2D tensor)
    asm volatile("tensor_load_to_lds %0, %1, %2, %3"
                 :: "s"(g0), "s"(g1), "s"(gz), "s"(gz)
                 : "memory");
}

// ============================================================
// Kernel 0: copy mesh_feats into out[:, :, 0:E]  (float4 vectorized)
// ============================================================
__global__ __launch_bounds__(256) void copy_mesh_kernel(const float* __restrict__ mesh,
                                                        float* __restrict__ out) {
    const int total = Bc * NQ * (Ec / 4);
    int idx = blockIdx.x * blockDim.x + threadIdx.x;
    if (idx >= total) return;
    int row = idx / (Ec / 4);
    int c   = idx - row * (Ec / 4);
    ((float4*)out)[(size_t)row * (2 * Ec / 4) + c] = ((const float4*)mesh)[idx];
}

// ============================================================
// Kernel 1: projection GEMM  Y = (X @ W^T + b)  -> f16
//   X:[M,K] f32, W:[N,K] f32.
//   TR=false: Y row-major [M,N] (scalar b16 stores)
//   TR=true : Y = per-batch transposed [B][N][NK] (one b128 store of
//             8 consecutive rows per lane) -- used for V.
// ============================================================
template <bool TR>
__global__ __launch_bounds__(256) void proj_gemm_kernel(const float* __restrict__ X,
                                                        const float* __restrict__ W,
                                                        const float* __restrict__ bias,
                                                        _Float16* __restrict__ Y,
                                                        int M, int N, int K) {
    const int wave = threadIdx.x >> 5;
    const int lane = threadIdx.x & 31;
    const int r    = lane & 15;       // row (A) / col (B) within tile
    const int hi   = lane >> 4;       // K-half select

    const int m0 = blockIdx.x * 16;
    const int n0 = (blockIdx.y * 8 + wave) * 16;

    const float* xp = X + (size_t)(m0 + r) * K + hi * 16;
    const float* wp = W + (size_t)(n0 + r) * K + hi * 16;

    v8f acc = {};
    for (int k0 = 0; k0 < K; k0 += 32) {
        v16h a = cvt16(xp + k0);
        v16h b = cvt16(wp + k0);
        acc = wmma_f16(a, b, acc);
    }

    const float bb = bias[n0 + r];
    if (TR) {
        // rows m0+8*hi .. +7 are consecutive -> pack 8 halfs, single 16B store
        const int bb_idx = m0 / NK;               // batch
        const int key0   = (m0 & (NK - 1)) + 8 * hi;
        union { _Float16 h[8]; uint4 u; } pk;
#pragma unroll
        for (int i = 0; i < 8; ++i) pk.h[i] = (_Float16)(acc[i] + bb);
        *(uint4*)&Y[((size_t)(bb_idx * Ec + n0 + r)) * NK + key0] = pk.u;
    } else {
#pragma unroll
        for (int i = 0; i < 8; ++i) {
            int row = m0 + i + 8 * hi;            // C layout: vgpr i -> M=i (+8 hi)
            Y[(size_t)row * N + n0 + r] = (_Float16)(acc[i] + bb);
        }
    }
}

// ============================================================
// Kernel 2: flash attention, one wave = one [16 x HD] query tile of one head.
//   qh:[B,NQ,E] f16   kh:[B,NK,E] f16   vT:[B,E,NK] f16   ctxh:[B,NQ,E] f16
//   K tiles arrive via double-buffered TDM loads into LDS (TENSORcnt).
//   4 waves / block (LDS: 4 waves x 2 x 4KB K buffers + 4 x 1KB P = 36KB).
// ============================================================
__global__ __launch_bounds__(128) void flash_attn_kernel(const _Float16* __restrict__ qh,
                                                         const _Float16* __restrict__ kh,
                                                         const _Float16* __restrict__ vT,
                                                         const int* __restrict__ lengths,
                                                         _Float16* __restrict__ ctxh) {
    __shared__ __align__(128) _Float16 Kl[4][2][32 * 64]; // K tile double buffer
    __shared__ __align__(128) _Float16 Pl[4][16 * 32];    // P transpose staging

    const int wave = threadIdx.x >> 5;
    const int lane = threadIdx.x & 31;
    const int r    = lane & 15;
    const int hi   = lane >> 4;

    const int wid = blockIdx.x * 4 + wave;          // 0 .. B*H*(NQ/16)-1
    const int qt  = wid & (NQ / 16 - 1);
    const int h   = (wid >> 7) & (Hc - 1);
    const int b   = wid >> 9;

    const int len = lengths[b];
    const int NCH = NK / 32;

    // ---- Q tile (16 x 64) as two K=32 A-operand slabs ----
    const _Float16* qp = qh + ((size_t)(b * NQ + qt * 16 + r)) * Ec + h * HD + hi * 16;
    const v16h aq0 = *(const v16h*)qp;
    const v16h aq1 = *(const v16h*)(qp + 32);

    // ---- per-head base pointers ----
    const _Float16* kbase = kh + ((size_t)b * NK) * Ec + h * HD;
    const _Float16* vrow[4];
#pragma unroll
    for (int t = 0; t < 4; ++t)
        vrow[t] = vT + ((size_t)(b * Ec + h * HD + t * 16 + r)) * NK;

    const unsigned kl_off0 = (unsigned)(uintptr_t)&Kl[wave][0][0];
    const unsigned kl_off1 = (unsigned)(uintptr_t)&Kl[wave][1][0];

    float m_run[8], l_run[8];
    v8f o0 = {}, o1 = {}, o2 = {}, o3 = {};
#pragma unroll
    for (int i = 0; i < 8; ++i) { m_run[i] = -1e30f; l_run[i] = 0.f; }

    const float SC = 0.125f * LOG2E;    // 1/sqrt(HD), log2 domain

    // prologue: start TDM for chunk 0
    tdm_load_k_tile(kbase, kl_off0);

    for (int kc = 0; kc < NCH; ++kc) {
        const int buf = kc & 1;
        const _Float16* Kt = &Kl[wave][buf][0];

        if (kc + 1 < NCH) {
            // buffer we are about to refill was last READ in the previous chunk (kc-1);
            // make sure those ds loads retired before TDM overwrites it
            asm volatile("s_wait_dscnt 0" ::: "memory");
            tdm_load_k_tile(kbase + (size_t)(kc + 1) * 32 * Ec,
                            buf ? kl_off0 : kl_off1);
            __builtin_amdgcn_s_wait_tensorcnt(1);   // oldest (current) landed
        } else {
            __builtin_amdgcn_s_wait_tensorcnt(0);
        }

        // prefetch next vT lines
        if (kc + 2 < NCH) {
#pragma unroll
            for (int t = 0; t < 4; ++t)
                __builtin_prefetch(vrow[t] + (kc + 2) * 32, 0, 1);
        }

        // ---- S = Q K^T from LDS K tile ----
        v8f s0 = {}, s1 = {};
        {
            v16h b0 = *(const v16h*)&Kt[(size_t)r * 64 + hi * 16];
            v16h b1 = *(const v16h*)&Kt[(size_t)r * 64 + hi * 16 + 32];
            s0 = wmma_f16(aq0, b0, s0);
            s0 = wmma_f16(aq1, b1, s0);
        }
        {
            v16h b0 = *(const v16h*)&Kt[(size_t)(16 + r) * 64 + hi * 16];
            v16h b1 = *(const v16h*)&Kt[(size_t)(16 + r) * 64 + hi * 16 + 32];
            s1 = wmma_f16(aq0, b0, s1);
            s1 = wmma_f16(aq1, b1, s1);
        }

        // ---- online softmax (log2 domain) with key-padding mask ----
        const int  col0 = kc * 32 + r;
        const bool ok0 = col0 < len, ok1 = (col0 + 16) < len;

#pragma unroll
        for (int i = 0; i < 8; ++i) {
            float a0 = ok0 ? s0[i] * SC : -1e30f;
            float a1 = ok1 ? s1[i] * SC : -1e30f;
            float rm = fmaxf(a0, a1);
#pragma unroll
            for (int msk = 1; msk < 16; msk <<= 1)
                rm = fmaxf(rm, __shfl_xor(rm, msk, 32));
            float mnew = fmaxf(m_run[i], rm);
            float sf   = exp2f(m_run[i] - mnew);
            m_run[i]   = mnew;
            float p0 = ok0 ? exp2f(a0 - mnew) : 0.f;
            float p1 = ok1 ? exp2f(a1 - mnew) : 0.f;
            float rs = p0 + p1;
#pragma unroll
            for (int msk = 1; msk < 16; msk <<= 1)
                rs += __shfl_xor(rs, msk, 32);
            l_run[i] = l_run[i] * sf + rs;
            o0[i] *= sf; o1[i] *= sf; o2[i] *= sf; o3[i] *= sf;
            // scatter P (C layout -> row major) into LDS
            int prow = i + 8 * hi;
            Pl[wave][prow * 32 + r]      = (_Float16)p0;
            Pl[wave][prow * 32 + 16 + r] = (_Float16)p1;
        }

        __builtin_amdgcn_wave_barrier();
        asm volatile("s_wait_dscnt 0" ::: "memory");   // wave-local transpose fence

        // ---- reload P as A operand (16 x 32) ----
        v16h ap = *(const v16h*)&Pl[wave][r * 32 + hi * 16];

        // ---- O += P V : B operand = contiguous keys from transposed V ----
        {
            v16h bv0 = *(const v16h*)&vrow[0][kc * 32 + hi * 16];
            v16h bv1 = *(const v16h*)&vrow[1][kc * 32 + hi * 16];
            v16h bv2 = *(const v16h*)&vrow[2][kc * 32 + hi * 16];
            v16h bv3 = *(const v16h*)&vrow[3][kc * 32 + hi * 16];
            o0 = wmma_f16(ap, bv0, o0);
            o1 = wmma_f16(ap, bv1, o1);
            o2 = wmma_f16(ap, bv2, o2);
            o3 = wmma_f16(ap, bv3, o3);
        }
        __builtin_amdgcn_wave_barrier();
    }

    // ---- normalize and store ctx (f16) ----
#pragma unroll
    for (int i = 0; i < 8; ++i) {
        float inv = 1.0f / l_run[i];
        int row = qt * 16 + i + 8 * hi;
        _Float16* cp = ctxh + ((size_t)(b * NQ + row)) * Ec + h * HD + r;
        cp[0]  = (_Float16)(o0[i] * inv);
        cp[16] = (_Float16)(o1[i] * inv);
        cp[32] = (_Float16)(o2[i] * inv);
        cp[48] = (_Float16)(o3[i] * inv);
    }
}

// ============================================================
// Kernel 3: output GEMM  mix = ctx @ Wo^T + bo -> out[:, :, E:2E] (f32)
// ============================================================
__global__ __launch_bounds__(256) void out_gemm_kernel(const _Float16* __restrict__ Ah,
                                                       const float* __restrict__ W,
                                                       const float* __restrict__ bias,
                                                       float* __restrict__ out) {
    const int K = Ec;
    const int wave = threadIdx.x >> 5;
    const int lane = threadIdx.x & 31;
    const int r    = lane & 15;
    const int hi   = lane >> 4;

    const int m0 = blockIdx.x * 16;
    const int n0 = (blockIdx.y * 8 + wave) * 16;

    const _Float16* ap = Ah + (size_t)(m0 + r) * K + hi * 16;
    const float*    wp = W  + (size_t)(n0 + r) * K + hi * 16;

    v8f acc = {};
    for (int k0 = 0; k0 < K; k0 += 32) {
        v16h a = *(const v16h*)(ap + k0);
        v16h b = cvt16(wp + k0);
        acc = wmma_f16(a, b, acc);
    }

    const float bb = bias[n0 + r];
#pragma unroll
    for (int i = 0; i < 8; ++i) {
        int row = m0 + i + 8 * hi;
        out[(size_t)row * (2 * Ec) + Ec + n0 + r] = acc[i] + bb;
    }
}

// ============================================================
extern "C" void kernel_launch(void* const* d_in, const int* in_sizes, int n_in,
                              void* d_out, int out_size, void* d_ws, size_t ws_size,
                              hipStream_t stream) {
    const float* mesh = (const float*)d_in[0];
    const float* pc   = (const float*)d_in[1];
    const float* Wq   = (const float*)d_in[2];
    const float* Wk   = (const float*)d_in[3];
    const float* Wv   = (const float*)d_in[4];
    const float* bq   = (const float*)d_in[5];
    const float* bk   = (const float*)d_in[6];
    const float* bv   = (const float*)d_in[7];
    const float* Wo   = (const float*)d_in[8];
    const float* bo   = (const float*)d_in[9];
    const int*   len  = (const int*)d_in[10];

    float* out = (float*)d_out;

    // workspace (f16): qh 4MB | kh 8MB | vT 8MB | ctxh 4MB
    _Float16* qh   = (_Float16*)d_ws;
    _Float16* kh   = qh + (size_t)Bc * NQ * Ec;
    _Float16* vT   = kh + (size_t)Bc * NK * Ec;     // [B][E][NK]
    _Float16* ctxh = vT + (size_t)Bc * NK * Ec;

    // mesh passthrough -> out[:, :, 0:E]
    {
        int n4 = Bc * NQ * Ec / 4;
        copy_mesh_kernel<<<(n4 + 255) / 256, 256, 0, stream>>>(mesh, out);
    }

    // projections (also the f32 -> f16 conversion point)
    proj_gemm_kernel<false><<<dim3(Bc * NQ / 16, Ec / 128), 256, 0, stream>>>(
        mesh, Wq, bq, qh, Bc * NQ, Ec, Ec);
    proj_gemm_kernel<false><<<dim3(Bc * NK / 16, Ec / 128), 256, 0, stream>>>(
        pc, Wk, bk, kh, Bc * NK, Ec, DPC);
    proj_gemm_kernel<true><<<dim3(Bc * NK / 16, Ec / 128), 256, 0, stream>>>(
        pc, Wv, bv, vT, Bc * NK, Ec, DPC);

    // flash attention: B*H*(NQ/16) = 2048 waves, 4 per block
    flash_attn_kernel<<<Bc * Hc * (NQ / 16) / 4, 128, 0, stream>>>(
        qh, kh, vT, len, ctxh);

    // output projection into out[:, :, E:2E]
    out_gemm_kernel<<<dim3(Bc * NQ / 16, Ec / 128), 256, 0, stream>>>(
        ctxh, Wo, bo, out);
}
